// Rewc_GIN_20194936226188
// MI455X (gfx1250) — matrix operand order repaired and checked by reading, not compiled, not run on hardware
//
#include <hip/hip_runtime.h>
#include <hip/hip_bf16.h>

// ---- problem constants (from reference) ----
#define NN 100000
#define EE 1200000
#define FF 128
#define DD 64
#define BB 128
#define CC 10
#define BN_EPS 1e-5f

typedef __bf16 bf16_t;
typedef __attribute__((ext_vector_type(16))) __bf16 v16bf;
typedef __attribute__((ext_vector_type(8)))  float  v8f;

__device__ inline bf16_t f2bf(float f) {
  union { float f; unsigned u; } v; v.f = f;
  unsigned r = (v.u + 0x7FFFu + ((v.u >> 16) & 1u)) >> 16;  // RNE
  union { unsigned short s; bf16_t b; } o; o.s = (unsigned short)r;
  return o.b;
}

__device__ inline v8f zero8() {
  v8f z;
#pragma unroll
  for (int i = 0; i < 8; ++i) z[i] = 0.f;
  return z;
}

// A-fragment: 16x32 bf16, M rows striped per lane (ISA 7.12.2 "16-bit A-Matrix 16x32").
// Per lane this is two contiguous 8-element (16B) chunks -> 2x ds_load_b128.
__device__ inline v16bf load_A_frag(const bf16_t* tile, int stride, int k0) {
  int lane = threadIdx.x & 31;
  int m  = lane & 15;
  int hi = lane >> 4;
  v16bf a;
#pragma unroll
  for (int e = 0; e < 16; ++e) {
    int vg = e >> 1, kk = e & 1;
    int k = (vg < 4) ? (vg * 2 + kk + hi * 8)
                     : (16 + (vg - 4) * 2 + kk + hi * 8);
    a[e] = tile[m * stride + k0 + k];
  }
  return a;
}

// Stage W[K x 64] (row-major fp32) into LDS pre-swizzled to B-fragment order:
// element e of fragment (kt, nt) for lane l lives at ((kt*4+nt)*32 + l)*16 + e,
// so a B-fragment load is one contiguous 32-byte v16bf read per lane.
template <int KT>
__device__ inline void stage_B_swizzled(const float* __restrict__ W,
                                        bf16_t* __restrict__ sW,
                                        int tid, int nthreads) {
  const int total = KT * 4 * 32 * 16;
  for (int i = tid; i < total; i += nthreads) {
    int e    = i & 15;
    int lane = (i >> 4) & 31;
    int nt   = (i >> 9) & 3;
    int kt   = i >> 11;
    int k = kt * 32 + (e & 7) + 8 * (lane >> 4) + 16 * (e >> 3);
    int n = nt * 16 + (lane & 15);
    sW[i] = f2bf(W[k * DD + n]);
  }
}

__device__ inline v16bf load_B_frag_sw(const bf16_t* sW, int kt, int nt) {
  int lane = threadIdx.x & 31;
  const v16bf* p = (const v16bf*)sW;
  return p[(kt * 4 + nt) * 32 + lane];
}

template <int KT>
__device__ inline v8f wmma_acc(const bf16_t* at, int astride,
                               const bf16_t* sW, int nt, v8f acc) {
#pragma unroll
  for (int kt = 0; kt < KT; ++kt) {
    v16bf a = load_A_frag(at, astride, kt * 32);
    v16bf b = load_B_frag_sw(sW, kt, nt);
    acc = __builtin_amdgcn_wmma_f32_16x16x32_bf16(false, a, false, b,
                                                  (short)0, acc, false, false);
  }
  return acc;
}

// ---------------- initial MLP: x[N,128] -> h0[N,64] ----------------
__global__ __launch_bounds__(128) void k_imlp(
    const float* __restrict__ x,
    const float* __restrict__ W1, const float* __restrict__ b1,
    const float* __restrict__ g1, const float* __restrict__ be1,
    const float* __restrict__ m1, const float* __restrict__ v1,
    const float* __restrict__ W2, const float* __restrict__ b2,
    const float* __restrict__ g2, const float* __restrict__ be2,
    const float* __restrict__ m2, const float* __restrict__ v2,
    float* __restrict__ h0)
{
  __shared__ __align__(32) bf16_t sW1[FF * DD];   // swizzled B-fragments, KT=4
  __shared__ __align__(32) bf16_t sW2[DD * DD];   // swizzled B-fragments, KT=2
  __shared__ __align__(32) bf16_t sX[4][16 * FF];
  __shared__ __align__(32) bf16_t sT[4][16 * DD];
  int tid = threadIdx.x;
  stage_B_swizzled<4>(W1, sW1, tid, 128);
  stage_B_swizzled<2>(W2, sW2, tid, 128);
  int wave = tid >> 5, lane = tid & 31;
  int nodeBase = blockIdx.x * 64 + wave * 16;
  bf16_t* xt = sX[wave];
  for (int i = lane; i < 16 * FF; i += 32) {
    int row = i >> 7, col = i & (FF - 1);
    int node = nodeBase + row;
    xt[i] = f2bf(node < NN ? x[(size_t)node * FF + col] : 0.f);
  }
  __syncthreads();

  int nlane = lane & 15, hi = lane >> 4;
  bf16_t* tt = sT[wave];
  // GEMM1 + bn1 affine + relu -> sT (bf16)
#pragma unroll
  for (int nt = 0; nt < 4; ++nt) {
    v8f acc = wmma_acc<4>(xt, FF, sW1, nt, zero8());
    int n = nt * 16 + nlane;
    float rs = rsqrtf(v1[n] + BN_EPS);
    float sc = rs * g1[n];
    float sh = (b1[n] - m1[n]) * sc + be1[n];
#pragma unroll
    for (int r = 0; r < 8; ++r)
      tt[(r + 8 * hi) * DD + n] = f2bf(fmaxf(acc[r] * sc + sh, 0.f));
  }
  __syncthreads();
  // GEMM2 + bn2 affine + relu -> h0
#pragma unroll
  for (int nt = 0; nt < 4; ++nt) {
    v8f acc = wmma_acc<2>(tt, DD, sW2, nt, zero8());
    int n = nt * 16 + nlane;
    float rs = rsqrtf(v2[n] + BN_EPS);
    float sc = rs * g2[n];
    float sh = (b2[n] - m2[n]) * sc + be2[n];
#pragma unroll
    for (int r = 0; r < 8; ++r) {
      int node = nodeBase + r + 8 * hi;
      if (node < NN) h0[(size_t)node * DD + n] = fmaxf(acc[r] * sc + sh, 0.f);
    }
  }
}

// ---------------- edge scatter: agg[dst] += h[src] where ew==layer ----------------
__global__ __launch_bounds__(256) void k_scatter(
    const float* __restrict__ h, const int* __restrict__ ei,
    const int* __restrict__ ew, float* __restrict__ agg, int layer)
{
  long gid = (long)blockIdx.x * 256 + threadIdx.x;
  int e = (int)(gid >> 4);
  if (e >= EE) return;
  if (ew[e] != layer) return;
  int f0 = ((int)gid & 15) * 4;
  int s = ei[e], d = ei[EE + e];
  const float4 v = *(const float4*)(h + (size_t)s * DD + f0);
  float* o = agg + (size_t)d * DD + f0;
  atomicAdd(o + 0, v.x); atomicAdd(o + 1, v.y);
  atomicAdd(o + 2, v.z); atomicAdd(o + 3, v.w);
}

// ---------------- GIN node update ----------------
__global__ __launch_bounds__(256) void k_ginnode(
    const float* __restrict__ h, const float* __restrict__ agg,
    const float* __restrict__ epsP,
    const float* __restrict__ ga, const float* __restrict__ ba,
    const float* __restrict__ ma, const float* __restrict__ va,
    const float* __restrict__ W1, const float* __restrict__ b1,
    const float* __restrict__ W2, const float* __restrict__ b2,
    const float* __restrict__ gb, const float* __restrict__ bb,
    const float* __restrict__ mb, const float* __restrict__ vb,
    float* __restrict__ hout)
{
  __shared__ __align__(32) bf16_t sW1[DD * DD];   // swizzled, KT=2
  __shared__ __align__(32) bf16_t sW2[DD * DD];   // swizzled, KT=2
  __shared__ __align__(32) bf16_t sZ[8][16 * DD];
  __shared__ __align__(32) bf16_t sT[8][16 * DD];
  int tid = threadIdx.x;
  stage_B_swizzled<2>(W1, sW1, tid, 256);
  stage_B_swizzled<2>(W2, sW2, tid, 256);
  float ep = 1.f + epsP[0];
  int wave = tid >> 5, lane = tid & 31;
  int nodeBase = blockIdx.x * 128 + wave * 16;
  bf16_t* zt = sZ[wave];
  for (int i = lane; i < 16 * DD; i += 32) {
    int row = i >> 6, col = i & 63;
    int node = nodeBase + row;
    float z = 0.f;
    if (node < NN) {
      size_t off = (size_t)node * DD + col;
      z = ep * h[off] + agg[off];
      z = (z - ma[col]) * rsqrtf(va[col] + BN_EPS) * ga[col] + ba[col];
      z = fmaxf(z, 0.f);
    }
    zt[i] = f2bf(z);
  }
  __syncthreads();

  int nlane = lane & 15, hi = lane >> 4;
  bf16_t* tt = sT[wave];
  // lin1 + relu
#pragma unroll
  for (int nt = 0; nt < 4; ++nt) {
    v8f acc = wmma_acc<2>(zt, DD, sW1, nt, zero8());
    int n = nt * 16 + nlane;
    float sh = b1[n];
#pragma unroll
    for (int r = 0; r < 8; ++r)
      tt[(r + 8 * hi) * DD + n] = f2bf(fmaxf(acc[r] + sh, 0.f));
  }
  __syncthreads();
  // lin2 + bn_b + relu
#pragma unroll
  for (int nt = 0; nt < 4; ++nt) {
    v8f acc = wmma_acc<2>(tt, DD, sW2, nt, zero8());
    int n = nt * 16 + nlane;
    float rs = rsqrtf(vb[n] + BN_EPS);
    float sc = rs * gb[n];
    float sh = (b2[n] - mb[n]) * sc + bb[n];
#pragma unroll
    for (int r = 0; r < 8; ++r) {
      int node = nodeBase + r + 8 * hi;
      if (node < NN) hout[(size_t)node * DD + n] = fmaxf(acc[r] * sc + sh, 0.f);
    }
  }
}

// ---------------- collapse conv stack + segment-sum pool ----------------
__global__ __launch_bounds__(256) void k_combine(
    const float* __restrict__ A0, const float* __restrict__ A1,
    const float* __restrict__ A2, const int* __restrict__ batch,
    const float* __restrict__ cw0, const float* __restrict__ cw1,
    const float* __restrict__ cw2,
    const float* __restrict__ cb0, const float* __restrict__ cb1,
    const float* __restrict__ cb2,
    const float* __restrict__ fw, const float* __restrict__ fb,
    float* __restrict__ pooled, float* __restrict__ counts)
{
  size_t idx = (size_t)blockIdx.x * 256 + threadIdx.x;
  if (idx >= (size_t)NN * DD) return;
  int n = (int)(idx >> 6), c = (int)(idx & 63);
  float f0 = fw[0], f1 = fw[1], f2v = fw[2];
  float a0 = f0 * cw0[0] + f1 * cw1[0] + f2v * cw2[0] + f0;
  float a1 = f1 * cw1[1] + f2v * cw2[1] + f1;
  float a2 = f2v * cw2[2] + f2v;
  float gc = f0 * cb0[0] + f1 * cb1[0] + f2v * cb2[0] + fb[0];
  float o = a0 * A0[idx] + a1 * A1[idx] + a2 * A2[idx] + gc;
  int b = batch[n];
  atomicAdd(&pooled[b * DD + c], o);
  if (c == 0) atomicAdd(&counts[b], 1.0f);
}

// ---------------- head: mean -> BN -> relu -> lin -> relu -> lin ----------------
__global__ __launch_bounds__(64) void k_head(
    const float* __restrict__ pooled, const float* __restrict__ counts,
    const float* __restrict__ g, const float* __restrict__ be,
    const float* __restrict__ mn, const float* __restrict__ vr,
    const float* __restrict__ W1, const float* __restrict__ b1,
    const float* __restrict__ W2, const float* __restrict__ b2,
    float* __restrict__ out)
{
  __shared__ float s1[DD];
  __shared__ float s2[DD];
  int r = blockIdx.x, c = threadIdx.x;
  float cnt = fmaxf(counts[r], 1.0f);
  float p = pooled[r * DD + c] / cnt;
  p = (p - mn[c]) * rsqrtf(vr[c] + BN_EPS) * g[c] + be[c];
  s1[c] = fmaxf(p, 0.f);
  __syncthreads();
  float acc = b1[c];
#pragma unroll 8
  for (int k = 0; k < DD; ++k) acc += s1[k] * W1[k * DD + c];
  s2[c] = fmaxf(acc, 0.f);
  __syncthreads();
  if (c < CC) {
    float a2 = b2[c];
#pragma unroll 8
    for (int j = 0; j < DD; ++j) a2 += s2[j] * W2[j * CC + c];
    out[r * CC + c] = a2;
  }
}

extern "C" void kernel_launch(void* const* d_in, const int* in_sizes, int n_in,
                              void* d_out, int out_size, void* d_ws, size_t ws_size,
                              hipStream_t stream) {
  (void)in_sizes; (void)n_in; (void)out_size; (void)ws_size;
  const float* x     = (const float*)d_in[0];
  const int*   ei    = (const int*)d_in[1];
  const int*   ew    = (const int*)d_in[2];
  const int*   batch = (const int*)d_in[3];
#define P(i) ((const float*)d_in[(i)])

  size_t nh = (size_t)NN * DD;
  float* ws     = (float*)d_ws;
  float* h0     = ws;
  float* h1     = h0 + nh;
  float* h2     = h1 + nh;
  float* agg    = h2 + nh;
  float* pooled = agg + nh;
  float* counts = pooled + (size_t)BB * DD;

  // initial MLP
  k_imlp<<<(NN + 63) / 64, 128, 0, stream>>>(
      x, P(4), P(5), P(6), P(7), P(8), P(9),
      P(10), P(11), P(12), P(13), P(14), P(15), h0);

  // GIN layer 1 (operates on h0; edges with weight==1)
  hipMemsetAsync(agg, 0, nh * sizeof(float), stream);
  k_scatter<<<(int)(((size_t)EE * 16 + 255) / 256), 256, 0, stream>>>(h0, ei, ew, agg, 1);
  k_ginnode<<<(NN + 127) / 128, 256, 0, stream>>>(
      h0, agg, P(16), P(17), P(18), P(19), P(20),
      P(21), P(22), P(23), P(24), P(25), P(26), P(27), P(28), h1);

  // GIN layer 2 (reference never reassigns h: still h0; edges with weight==2)
  hipMemsetAsync(agg, 0, nh * sizeof(float), stream);
  k_scatter<<<(int)(((size_t)EE * 16 + 255) / 256), 256, 0, stream>>>(h0, ei, ew, agg, 2);
  k_ginnode<<<(NN + 127) / 128, 256, 0, stream>>>(
      h0, agg, P(29), P(30), P(31), P(32), P(33),
      P(34), P(35), P(36), P(37), P(38), P(39), P(40), P(41), h2);

  // collapse conv stack + pool
  hipMemsetAsync(pooled, 0, ((size_t)BB * DD + BB) * sizeof(float), stream);
  k_combine<<<(int)(((size_t)NN * DD + 255) / 256), 256, 0, stream>>>(
      h0, h1, h2, batch,
      P(42), P(43), P(44), P(45), P(46), P(47), P(48), P(49),
      pooled, counts);

  // head
  k_head<<<BB, 64, 0, stream>>>(
      pooled, counts, P(50), P(51), P(52), P(53),
      P(54), P(55), P(56), P(57), (float*)d_out);
#undef P
}
